// NT_Xent_Loss_4002909520385
// MI455X (gfx1250) — matrix operand order repaired and checked by reading, not compiled, not run on hardware
//
#include <hip/hip_runtime.h>

typedef _Float16 v16h __attribute__((ext_vector_type(16)));
typedef _Float16 v8h  __attribute__((ext_vector_type(8)));
typedef float    v8f  __attribute__((ext_vector_type(8)));

#define B_ROWS 4096
#define DIM    256
#define NROWS  8192              // 2B
#define INV_T  10.0f             // 1 / 0.1
#define LDS_STRIDE (DIM + 8)     // 264 halves: 16B-aligned, bank-conflict-free
#define COL_SPLIT  8
#define TILES_PER_BLOCK (NROWS / 16 / COL_SPLIT)   // 64

#if defined(__HIP_DEVICE_COMPILE__) && __has_builtin(__builtin_amdgcn_global_load_async_to_lds_b128)
#define USE_ASYNC_LDS 1
#else
#define USE_ASYNC_LDS 0
#endif

#if USE_ASYNC_LDS
// Prototype (from clang diagnostic): param 0 is `int __vector(4) __device__ *`
// (AS1), param 1 is the LDS-side pointer (AS3), then i32 offset, i32 cpol.
typedef int async_v4i __attribute__((vector_size(16)));
typedef __attribute__((address_space(1))) async_v4i g_v4i;
typedef __attribute__((address_space(3))) async_v4i l_v4i;
#endif

// ---------------------------------------------------------------------------
// Kernel 1: L2-normalize rows of [z_i ; z_j] -> f16 matrix R [8192 x 256]
// ---------------------------------------------------------------------------
__global__ void __launch_bounds__(256)
k_prep(const float* __restrict__ zi, const float* __restrict__ zj,
       _Float16* __restrict__ Rh) {
    const int row  = (int)((blockIdx.x * blockDim.x + threadIdx.x) >> 5);
    const int lane = (int)(threadIdx.x & 31);
    if (row >= NROWS) return;

    const float* src = (row < B_ROWS) ? (zi + (size_t)row * DIM)
                                      : (zj + (size_t)(row - B_ROWS) * DIM);
    const float4* s4 = (const float4*)(src + lane * 8);
    float4 a = s4[0];
    float4 b = s4[1];
    float ss = a.x*a.x + a.y*a.y + a.z*a.z + a.w*a.w
             + b.x*b.x + b.y*b.y + b.z*b.z + b.w*b.w;
    #pragma unroll
    for (int off = 16; off >= 1; off >>= 1)
        ss += __shfl_xor(ss, off, 32);
    const float inv = 1.0f / fmaxf(sqrtf(ss), 1e-12f);

    _Float16* dst = Rh + (size_t)row * DIM + lane * 8;
    dst[0] = (_Float16)(a.x * inv); dst[1] = (_Float16)(a.y * inv);
    dst[2] = (_Float16)(a.z * inv); dst[3] = (_Float16)(a.w * inv);
    dst[4] = (_Float16)(b.x * inv); dst[5] = (_Float16)(b.y * inv);
    dst[6] = (_Float16)(b.z * inv); dst[7] = (_Float16)(b.w * inv);
}

// ---------------------------------------------------------------------------
// Kernel 2: pos[k] = dot(R[k], R[k+B]) in f32 (consistent with WMMA sim).
// ---------------------------------------------------------------------------
__global__ void __launch_bounds__(256)
k_pos(const _Float16* __restrict__ Rh, float* __restrict__ pos) {
    const int k    = (int)((blockIdx.x * blockDim.x + threadIdx.x) >> 5);
    const int lane = (int)(threadIdx.x & 31);
    if (k >= B_ROWS) return;

    const _Float16* a = Rh + (size_t)k * DIM + lane * 8;
    const _Float16* b = Rh + (size_t)(k + B_ROWS) * DIM + lane * 8;
    float s = 0.0f;
    #pragma unroll
    for (int i = 0; i < 8; ++i)
        s += (float)a[i] * (float)b[i];
    #pragma unroll
    for (int off = 16; off >= 1; off >>= 1)
        s += __shfl_xor(s, off, 32);
    if (lane == 0) pos[k] = s;
}

// ---------------------------------------------------------------------------
// Async (or fallback) cooperative load of one 16-col x 256-K B tile into LDS.
// 256 threads x 32 bytes each = 8 KB tile (padded rows in LDS).
// ---------------------------------------------------------------------------
__device__ __forceinline__ void load_tile(const _Float16* __restrict__ Rh,
                                          _Float16* shDst, int colBase, int tid) {
    const int idx = tid * 16;            // half index within the 16x256 tile
    const int col = idx >> 8;            // 0..15
    const int k   = idx & (DIM - 1);     // 0..255, multiple of 16
    const _Float16* gsrc = Rh + (size_t)(colBase + col) * DIM + k;
    _Float16* dst = shDst + col * LDS_STRIDE + k;
#if USE_ASYNC_LDS
    __builtin_amdgcn_global_load_async_to_lds_b128((g_v4i*)gsrc,
                                                   (l_v4i*)dst, 0, 0);
    __builtin_amdgcn_global_load_async_to_lds_b128((g_v4i*)(gsrc + 8),
                                                   (l_v4i*)(dst + 8), 0, 0);
#else
    v8h r0 = *(const v8h*)gsrc;
    v8h r1 = *(const v8h*)(gsrc + 8);
    *(v8h*)dst       = r0;
    *(v8h*)(dst + 8) = r1;
#endif
}

__device__ __forceinline__ void async_wait_le(int n) {
#if USE_ASYNC_LDS
#if __has_builtin(__builtin_amdgcn_s_wait_asynccnt)
    if (n == 0) __builtin_amdgcn_s_wait_asynccnt(0);
    else        __builtin_amdgcn_s_wait_asynccnt(2);
#else
    if (n == 0) asm volatile("s_wait_asynccnt 0x0" ::: "memory");
    else        asm volatile("s_wait_asynccnt 0x2" ::: "memory");
#endif
#else
    (void)n;
#endif
}

// ---------------------------------------------------------------------------
// Kernel 3: fused Gram-row exp-sum with LDS-shared, double-buffered B tiles.
// grid = (64 row-blocks of 128 rows, 8 column chunks); block = 256 (8 waves).
// Wave w owns rows [blk.x*128 + 16w, +16); all 8 waves share each B tile.
// ---------------------------------------------------------------------------
__global__ void __launch_bounds__(256)
k_simrows(const _Float16* __restrict__ Rh, float* __restrict__ part) {
    __shared__ _Float16 sh[2][16 * LDS_STRIDE];

    const int tid  = (int)threadIdx.x;
    const int lane = tid & 31;
    const int wv   = tid >> 5;               // wave id -> row sub-block
    const int g    = lane >> 4;              // half-wave select
    const int mn   = lane & 15;              // A row-in-tile / B col-in-tile

    const int rowBase = (int)blockIdx.x * 128 + wv * 16;
    const int tile0   = (int)blockIdx.y * TILES_PER_BLOCK;

    // Preload A fragments for this wave's 16 rows (K = 0..255).
    // 16-bit A 16x32 layout: lane(g,m) holds K in [8g,8g+8) and [16+8g,16+8g+8).
    const _Float16* arow = Rh + (size_t)(rowBase + mn) * DIM;
    v16h A[8];
    #pragma unroll
    for (int c = 0; c < 8; ++c) {
        v8h lo = *(const v8h*)(arow + 32 * c + 8 * g);
        v8h hi = *(const v8h*)(arow + 32 * c + 16 + 8 * g);
        A[c] = __builtin_shufflevector(lo, hi,
                 0,1,2,3,4,5,6,7,8,9,10,11,12,13,14,15);
    }

    float sums[8] = {0.f,0.f,0.f,0.f,0.f,0.f,0.f,0.f};

    load_tile(Rh, sh[0], tile0 * 16, tid);

    for (int t = 0; t < TILES_PER_BLOCK; ++t) {
        const int buf  = t & 1;
        const bool more = (t + 1) < TILES_PER_BLOCK;
        if (more) load_tile(Rh, sh[buf ^ 1], (tile0 + t + 1) * 16, tid);
        async_wait_le(more ? 2 : 0);   // tile t's async writes have landed
        __syncthreads();               // ...for every wave in the block

        // 16-bit B 32x16 layout: lane(g,n) holds K in [8g,8g+8) and [16+8g,16+8g+8) of column n.
        const _Float16* bcol = sh[buf] + mn * LDS_STRIDE;
        v8f acc0 = {};
        v8f acc1 = {};
        #pragma unroll
        for (int c = 0; c < 8; c += 2) {
            v8h lo0 = *(const v8h*)(bcol + 32 * c + 8 * g);
            v8h hi0 = *(const v8h*)(bcol + 32 * c + 16 + 8 * g);
            v16h B0 = __builtin_shufflevector(lo0, hi0,
                        0,1,2,3,4,5,6,7,8,9,10,11,12,13,14,15);
            acc0 = __builtin_amdgcn_wmma_f32_16x16x32_f16(
                       false, A[c], false, B0, (short)0, acc0, false, false);
            v8h lo1 = *(const v8h*)(bcol + 32 * (c + 1) + 8 * g);
            v8h hi1 = *(const v8h*)(bcol + 32 * (c + 1) + 16 + 8 * g);
            v16h B1 = __builtin_shufflevector(lo1, hi1,
                        0,1,2,3,4,5,6,7,8,9,10,11,12,13,14,15);
            acc1 = __builtin_amdgcn_wmma_f32_16x16x32_f16(
                       false, A[c + 1], false, B1, (short)0, acc1, false, false);
        }

        // Epilogue: D layout -> lane(g,n) VGPR r = sim[rowBase+8g+r][colBase+n]
        const int colG = (tile0 + t) * 16 + mn;
        #pragma unroll
        for (int r = 0; r < 8; ++r) {
            const int rowG = rowBase + 8 * g + r;
            const float e = __expf((acc0[r] + acc1[r]) * INV_T);
            sums[r] += (rowG != colG) ? e : 0.0f;   // diag: exp(-9e16) == 0
        }
        __syncthreads();               // done reading buf before it's reloaded
    }

    // Butterfly-reduce over the 16 lanes of each half-wave.
    #pragma unroll
    for (int off = 1; off <= 8; off <<= 1) {
        #pragma unroll
        for (int r = 0; r < 8; ++r)
            sums[r] += __shfl_xor(sums[r], off, 32);
    }

    if (mn == 0) {
        #pragma unroll
        for (int r = 0; r < 8; ++r)
            part[(size_t)(rowBase + 8 * g + r) * COL_SPLIT + blockIdx.y] = sums[r];
    }
}

// ---------------------------------------------------------------------------
// Kernel 4: deterministic final reduction:
// loss = mean_k( log(sum_c part[k][c]) - pos[k mod B] * 10 )
// ---------------------------------------------------------------------------
__global__ void __launch_bounds__(256)
k_final(const float* __restrict__ part, const float* __restrict__ pos,
        float* __restrict__ out) {
    __shared__ float red[256];
    float local = 0.0f;
    for (int k = (int)threadIdx.x; k < NROWS; k += 256) {
        float rs = 0.0f;
        #pragma unroll
        for (int c = 0; c < COL_SPLIT; ++c)
            rs += part[(size_t)k * COL_SPLIT + c];
        const float p = pos[(k < B_ROWS) ? k : (k - B_ROWS)];
        local += (logf(rs) - p * INV_T);
    }
    red[threadIdx.x] = local;
    __syncthreads();
    for (int s = 128; s > 0; s >>= 1) {
        if ((int)threadIdx.x < s) red[threadIdx.x] += red[threadIdx.x + s];
        __syncthreads();
    }
    if (threadIdx.x == 0) out[0] = red[0] / (float)NROWS;
}

// ---------------------------------------------------------------------------
extern "C" void kernel_launch(void* const* d_in, const int* in_sizes, int n_in,
                              void* d_out, int out_size, void* d_ws, size_t ws_size,
                              hipStream_t stream) {
    (void)in_sizes; (void)n_in; (void)out_size; (void)ws_size;
    const float* zi = (const float*)d_in[0];
    const float* zj = (const float*)d_in[1];
    float* out = (float*)d_out;

    char* ws = (char*)d_ws;
    _Float16* Rh  = (_Float16*)ws;                                  // 4 MB
    float*    pos = (float*)(ws + (size_t)NROWS * DIM * 2);         // 16 KB
    float*    prt = (float*)(ws + (size_t)NROWS * DIM * 2
                                + (size_t)B_ROWS * 4);              // 256 KB

    k_prep   <<<NROWS / 8,  256, 0, stream>>>(zi, zj, Rh);
    k_pos    <<<B_ROWS / 8, 256, 0, stream>>>(Rh, pos);
    k_simrows<<<dim3(NROWS / 128, COL_SPLIT), 256, 0, stream>>>(Rh, prt);
    k_final  <<<1,          256, 0, stream>>>(prt, pos, out);
}
